// GAT_3401614098764
// MI455X (gfx1250) — matrix operand order repaired and checked by reading, not compiled, not run on hardware
//
#include <hip/hip_runtime.h>
#include <hip/hip_bf16.h>

typedef __attribute__((ext_vector_type(16))) _Float16 v16h;
typedef __attribute__((ext_vector_type(8)))  _Float16 v8h;
typedef __attribute__((ext_vector_type(8)))  float    v8f;

#define N_NODES 50000
#define N_EDGES 800000
#define E_TOT   850000          // edges + self-loops
#define KD      256             // contraction dim for all three layers
#define NEG_SLOPE 0.2f
#define CDIV(a,b) (((a)+(b)-1)/(b))

// ---------- helpers ----------

// Order-preserving float -> uint mapping so we can use hardware atomicMax(u32)
__device__ __forceinline__ unsigned fkey(float x){
  unsigned u = __float_as_uint(x);
  return (u & 0x80000000u) ? ~u : (u | 0x80000000u);
}
__device__ __forceinline__ float funkey(unsigned k){
  unsigned u = (k & 0x80000000u) ? (k & 0x7FFFFFFFu) : ~k;
  return __uint_as_float(u);
}
// edge_index is [2, N_EDGES] int64; self-loops appended virtually at e >= N_EDGES
__device__ __forceinline__ void edge_sd(const long long* __restrict__ ei, int e, int& s, int& d){
  if (e < N_EDGES){ s = (int)ei[e]; d = (int)ei[N_EDGES + e]; }
  else { s = e - N_EDGES; d = s; }
}

// ---------- kernels ----------

// f32 -> f16 (optionally fused ReLU) for WMMA A operand
__global__ void k_cvt_f16(const float* __restrict__ in, _Float16* __restrict__ out,
                          int n, int relu){
  int i = blockIdx.x*blockDim.x + threadIdx.x;
  if (i < n){
    float v = in[i];
    if (relu && v < 0.f) v = 0.f;
    out[i] = (_Float16)v;
  }
}

// Pack W[K,M] (f32) into per-lane WMMA B-fragment order (f16):
// Bpack[((tileC*(KD/32) + kidx)*32 + lane)*16 + t] = W[kidx*32 + (lane>>4)*16 + t][tileC*16 + (lane&15)]
__global__ void k_pack_b(const float* __restrict__ W, _Float16* __restrict__ Bpack, int M){
  int idx = blockIdx.x*blockDim.x + threadIdx.x;
  if (idx >= KD*M) return;
  int t     = idx & 15;
  int lane  = (idx >> 4) & 31;
  int tmp   = idx >> 9;
  int kidx  = tmp & 7;            // KD/32 == 8
  int tileC = tmp >> 3;
  int k   = kidx*32 + ((lane >> 4) << 4) + t;
  int col = tileC*16 + (lane & 15);
  Bpack[idx] = (_Float16)W[(size_t)k*M + col];
}

// C[N,M] = A[N,256] @ B[256,M]; f16 in, f32 acc via v_wmma_f32_16x16x32_f16.
// One block (8 waves) per 16-row strip and 128-col group: the A tile (16x256 f16 = 8KB) is staged
// into LDS once per block by the Tensor Data Mover, then each wave computes one
// 16x16 output tile (one col-tile per wave), fully unrolled over K.
__global__ void __launch_bounds__(256)
k_gemm_wmma(const _Float16* __restrict__ A,
            const _Float16* __restrict__ Bpack,
            float* __restrict__ C,
            int M, int colGroups){
  __shared__ __align__(16) _Float16 As[16*KD];   // 8KB A tile

  int rowBlock = blockIdx.x / colGroups;
  int colGroup = blockIdx.x % colGroups;
  int waveId   = threadIdx.x >> 5;
  int lane     = threadIdx.x & 31;

  // Generic pointers into the LDS aperture carry the LDS byte offset in the
  // low 32 bits (ISA 10.2). Passing this into the asm also *captures* As so
  // the compiler must treat the TDM op as a writer of the LDS tile.
  unsigned ldsAddr = (unsigned)(size_t)(&As[0]);

  // ---- TDM: stage A[rowBlock*16 .. +16, 0..256) -> LDS (wave 0 only) ----
  if (threadIdx.x < 32){
    unsigned long long ga = (unsigned long long)(A + (size_t)rowBlock * 16 * KD);
    unsigned g0w0 = 1u;                                  // count=1, user descriptor
    unsigned g0w1 = ldsAddr;                             // lds_addr (byte offset of As)
    unsigned g0w2 = (unsigned)(ga & 0xFFFFFFFFull);      // global_addr[31:0]
    unsigned g0w3 = (unsigned)((ga >> 32) & 0x01FFFFFFull) | 0x80000000u; // addr hi | type=2
    unsigned g1w0 = 0x10000u;                            // data_size=1 (2B), no multicast/pad
    unsigned g1w1 = (KD & 0xFFFFu) << 16;                // tensor_dim0 lo16
    unsigned g1w2 = (KD >> 16) | (16u << 16);            // tensor_dim0 hi | tensor_dim1 lo
    unsigned g1w3 = (KD & 0xFFFFu) << 16;                // tensor_dim1 hi(0) | tile_dim0=256
    unsigned g1w4 = 16u;                                 // tile_dim1=16, tile_dim2=0
    unsigned g1w5 = (unsigned)KD;                        // tensor_dim0_stride lo32 = 256
    unsigned g1w6 = 0u;                                  // stride0 hi | dim1_stride lo
    unsigned g1w7 = 0u;                                  // dim1_stride hi
    asm volatile(
      "s_mov_b32 s64, %0\n\t"
      "s_mov_b32 s65, %1\n\t"
      "s_mov_b32 s66, %2\n\t"
      "s_mov_b32 s67, %3\n\t"
      "s_mov_b32 s72, %4\n\t"
      "s_mov_b32 s73, %5\n\t"
      "s_mov_b32 s74, %6\n\t"
      "s_mov_b32 s75, %7\n\t"
      "s_mov_b32 s76, %8\n\t"
      "s_mov_b32 s77, %9\n\t"
      "s_mov_b32 s78, %10\n\t"
      "s_mov_b32 s79, %11\n\t"
      "tensor_load_to_lds s[64:67], s[72:79]\n\t"
      :
      : "s"(g0w0), "s"(g0w1), "s"(g0w2), "s"(g0w3),
        "s"(g1w0), "s"(g1w1), "s"(g1w2), "s"(g1w3),
        "s"(g1w4), "s"(g1w5), "s"(g1w6), "s"(g1w7)
      : "s64","s65","s66","s67","s72","s73","s74","s75","s76","s77","s78","s79","memory");
    __builtin_amdgcn_s_wait_tensorcnt(0);
  }
  __syncthreads();

  int tileC = colGroup*8 + waveId;
  if (tileC*16 >= M) return;                 // wave-uniform: EXEC all-ones for WMMA

  int hi  = lane >> 4;
  int l15 = lane & 15;
  int akb = hi * 16;                         // A fragment K sub-offset (contiguous k = 16*hi + s, same as Bpack)
  int bcol = tileC*16 + l15;

  const _Float16* arow = As + l15*KD;
  const _Float16* bp   = Bpack + ((size_t)tileC*(KD/32)*32 + lane)*16;

  v8f acc = {};
#pragma unroll
  for (int kidx = 0; kidx < KD/32; ++kidx){
    int k0 = kidx*32;
    v8h alo = *(const v8h*)(arow + k0 + akb);        // ds_load_b128
    v8h ahi = *(const v8h*)(arow + k0 + 8 + akb);    // ds_load_b128
    v16h a = __builtin_shufflevector(alo, ahi, 0,1,2,3,4,5,6,7,8,9,10,11,12,13,14,15);
    v16h b = *(const v16h*)(bp + (size_t)kidx*32*16); // 32B contiguous per lane
    acc = __builtin_amdgcn_wmma_f32_16x16x32_f16(
        false, a, false, b, (short)0, acc, false, false);
  }
#pragma unroll
  for (int i = 0; i < 8; ++i){
    int m = hi*8 + i;
    C[(size_t)(rowBlock*16 + m) * M + bcol] = acc[i];
  }
}

// a_s[n,h] = sum_c h[n,h,c]*att_src[h,c]; a_d likewise
__global__ void k_att(const float* __restrict__ H, const float* __restrict__ att_s,
                      const float* __restrict__ att_d, float* __restrict__ as_,
                      float* __restrict__ ad_, int Nn, int heads, int C){
  int i = blockIdx.x*blockDim.x + threadIdx.x;
  if (i >= Nn*heads) return;
  int n = i / heads, h = i % heads;
  const float* hp = H + (size_t)n*heads*C + (size_t)h*C;
  const float* sp = att_s + h*C;
  const float* dp = att_d + h*C;
  float s = 0.f, d = 0.f;
  for (int c = 0; c < C; ++c){ float v = hp[c]; s += v*sp[c]; d += v*dp[c]; }
  as_[i] = s; ad_[i] = d;
}

__global__ void k_init_nodes(unsigned* __restrict__ mkey, float* __restrict__ denom, int n){
  int i = blockIdx.x*blockDim.x + threadIdx.x;
  if (i < n){ mkey[i] = 0x007FFFFFu; /* fkey(-inf) */ denom[i] = 0.f; }
}

__global__ void k_init_bias(float* __restrict__ out, const float* __restrict__ bias,
                            int n, int M){
  int i = blockIdx.x*blockDim.x + threadIdx.x;
  if (i < n) out[i] = bias[i % M];
}

// pass 1: e = leaky_relu(a_s[src] + a_d[dst]); segment max over dst via ordered-uint atomicMax
__global__ void k_edge_max(const long long* __restrict__ ei, const float* __restrict__ as_,
                           const float* __restrict__ ad_, float* __restrict__ eraw,
                           unsigned* __restrict__ mkey, int heads){
  int idx = blockIdx.x*blockDim.x + threadIdx.x;
  if (idx >= E_TOT*heads) return;
  int e = idx / heads, h = idx % heads;
  int s, d; edge_sd(ei, e, s, d);
  float v = as_[s*heads + h] + ad_[d*heads + h];
  v = (v > 0.f) ? v : NEG_SLOPE * v;
  eraw[idx] = v;
  atomicMax(&mkey[d*heads + h], fkey(v));
}

// pass 2: ex = exp(e - m[dst]); denom[dst] += ex
__global__ void k_edge_exp(const long long* __restrict__ ei, float* __restrict__ eraw,
                           const unsigned* __restrict__ mkey, float* __restrict__ denom,
                           int heads){
  int idx = blockIdx.x*blockDim.x + threadIdx.x;
  if (idx >= E_TOT*heads) return;
  int e = idx / heads, h = idx % heads;
  int s, d; edge_sd(ei, e, s, d);
  float m = funkey(mkey[d*heads + h]);
  float x = expf(eraw[idx] - m);
  eraw[idx] = x;
  atomicAdd(&denom[d*heads + h], x);
}

// pass 3: out[dst] += alpha * h[src]   (out pre-initialized with bias)
__global__ void k_edge_agg(const long long* __restrict__ ei, const float* __restrict__ ex,
                           const float* __restrict__ denom, const float* __restrict__ H,
                           float* __restrict__ out, int heads, int C){
  int idx = blockIdx.x*blockDim.x + threadIdx.x;
  if (idx >= E_TOT*heads) return;
  int e = idx / heads, h = idx % heads;
  int s, d; edge_sd(ei, e, s, d);
  float alpha = ex[idx] / (denom[d*heads + h] + 1e-16f);
  const float* hp = H   + (size_t)s*heads*C + (size_t)h*C;
  float*       op = out + (size_t)d*heads*C + (size_t)h*C;
  for (int c = 0; c < C; ++c) atomicAdd(&op[c], alpha * hp[c]);
}

// ---------- launch ----------

extern "C" void kernel_launch(void* const* d_in, const int* in_sizes, int n_in,
                              void* d_out, int out_size, void* d_ws, size_t ws_size,
                              hipStream_t stream){
  (void)in_sizes; (void)n_in; (void)out_size; (void)ws_size;
  const float*     x  = (const float*)d_in[0];
  const long long* ei = (const long long*)d_in[1];
  const float* W1 = (const float*)d_in[2];
  const float* as1 = (const float*)d_in[3];
  const float* ad1 = (const float*)d_in[4];
  const float* b1 = (const float*)d_in[5];
  const float* W2 = (const float*)d_in[6];
  const float* as2 = (const float*)d_in[7];
  const float* ad2 = (const float*)d_in[8];
  const float* b2 = (const float*)d_in[9];
  const float* W3 = (const float*)d_in[10];
  const float* as3 = (const float*)d_in[11];
  const float* ad3 = (const float*)d_in[12];
  const float* b3 = (const float*)d_in[13];

  // workspace carve-out (256B aligned)
  char* ws = (char*)d_ws; size_t off = 0;
  auto carve = [&](size_t bytes)->void*{
    void* p = ws + off; off += (bytes + 255) & ~(size_t)255; return p;
  };
  _Float16* Hf16  = (_Float16*)carve((size_t)N_NODES*KD*sizeof(_Float16));
  _Float16* Bpack = (_Float16*)carve((size_t)KD*256*sizeof(_Float16));
  float*    Hbuf  = (float*)carve((size_t)N_NODES*256*sizeof(float));   // gemm out (h)
  float*    AGG   = (float*)carve((size_t)N_NODES*256*sizeof(float));   // aggregated out
  float*    a_s   = (float*)carve((size_t)N_NODES*4*sizeof(float));
  float*    a_d   = (float*)carve((size_t)N_NODES*4*sizeof(float));
  unsigned* mkey  = (unsigned*)carve((size_t)N_NODES*4*sizeof(unsigned));
  float*    denom = (float*)carve((size_t)N_NODES*4*sizeof(float));
  float*    eraw  = (float*)carve((size_t)E_TOT*4*sizeof(float));

  const int T = 256;
  const int nFeat = N_NODES*KD;

  auto layer = [&](const float* inFeat, int relu_in, const float* W, int M,
                   const float* attS, const float* attD, const float* bias,
                   int heads, int C, float* outFeat){
    // operands to f16 (A converted, B packed into fragment order)
    k_cvt_f16<<<CDIV(nFeat,T), T, 0, stream>>>(inFeat, Hf16, nFeat, relu_in);
    k_pack_b <<<CDIV(KD*M,T), T, 0, stream>>>(W, Bpack, M);
    // GEMM: h = X @ W  (TDM-staged A tile + WMMA)
    int colGroups = CDIV(M/16, 8);
    int blocks = (N_NODES/16) * colGroups;
    k_gemm_wmma<<<blocks, T, 0, stream>>>(Hf16, Bpack, Hbuf, M, colGroups);
    // attention coefficients
    k_att<<<CDIV(N_NODES*heads,T), T, 0, stream>>>(Hbuf, attS, attD, a_s, a_d,
                                                   N_NODES, heads, C);
    // segment softmax + aggregate
    k_init_nodes<<<CDIV(N_NODES*heads,T), T, 0, stream>>>(mkey, denom, N_NODES*heads);
    k_init_bias<<<CDIV(N_NODES*M,T), T, 0, stream>>>(outFeat, bias, N_NODES*M, M);
    int EW = E_TOT*heads;
    k_edge_max<<<CDIV(EW,T), T, 0, stream>>>(ei, a_s, a_d, eraw, mkey, heads);
    k_edge_exp<<<CDIV(EW,T), T, 0, stream>>>(ei, eraw, mkey, denom, heads);
    k_edge_agg<<<CDIV(EW,T), T, 0, stream>>>(ei, eraw, denom, Hbuf, outFeat, heads, C);
  };

  // layer 1: x -> AGG ; layer 2: relu(AGG) -> AGG ; layer 3: relu(AGG) -> d_out
  layer(x,   0, W1, 256, as1, ad1, b1, 4, 64, AGG);
  layer(AGG, 1, W2, 256, as2, ad2, b2, 4, 64, AGG);
  layer(AGG, 1, W3, 16,  as3, ad3, b3, 1, 16, (float*)d_out);
}
